// Atten_Head_86912958201907
// MI455X (gfx1250) — matrix operand order repaired and checked by reading, not compiled, not run on hardware
//
#include <hip/hip_runtime.h>
#include <hip/hip_bf16.h>

// ---------------------------------------------------------------------------
// Attention head, CDNA5 (gfx1250, wave32, WMMA bf16 / f32-accum)
//   B=4, S=2048, D_IN=1024, D_OUT=64, causal mask
// Kernel 1: fused QKV projection (HBM-bound: ~100 MB f32 activations)
// Kernel 2: flash attention with online softmax; K/V staged to LDS with
//           GLOBAL_LOAD_ASYNC_TO_LDS_B128 (ASYNCcnt path, ISA 08_async_tensor)
// ---------------------------------------------------------------------------

typedef __attribute__((ext_vector_type(16))) __bf16 v16bf;
typedef __attribute__((ext_vector_type(8)))  __bf16 v8bf;
typedef __attribute__((ext_vector_type(8)))  float  v8f;

#define WMMA_BF16(a, b, c) \
  __builtin_amdgcn_wmma_f32_16x16x32_bf16(false, (a), false, (b), (short)0, (c), false, false)

static __device__ __forceinline__ v16bf cat16(v8bf lo, v8bf hi) {
  return __builtin_shufflevector(lo, hi, 0, 1, 2, 3, 4, 5, 6, 7,
                                        8, 9, 10, 11, 12, 13, 14, 15);
}
static __device__ __forceinline__ v8bf cvt8(v8f v) {
  return __builtin_convertvector(v, v8bf);
}

// Async global->LDS 16B copy (per-lane). VDST = LDS byte address (flat addr
// low 32 bits == wave-relative LDS offset per ISA 10.2), VADDR = 64-bit VA.
static __device__ __forceinline__ void async_copy_b128(void* lds_dst, const void* gsrc) {
  unsigned lds_off = (unsigned)(unsigned long long)lds_dst;
  unsigned long long va = (unsigned long long)gsrc;
  asm volatile("global_load_async_to_lds_b128 %0, %1, off"
               :: "v"(lds_off), "v"(va) : "memory");
}
static __device__ __forceinline__ void wait_asynccnt0() {
  asm volatile("s_wait_asynccnt 0x0" ::: "memory");
}

enum { BATCH = 4, SEQ = 2048, DIN = 1024, DOUT = 64, NROWS = BATCH * SEQ };

// ---------------------------------------------------------------------------
// Kernel 1: Y = X @ W for Q/K/V (blockIdx.y selects which), f32 in, bf16 out.
// Each wave: one 16(M) x 16(N) tile, K-loop 32 per v_wmma_f32_16x16x32_bf16.
// A-frag layout (16-bit A 16x32): lane%16 = M row; lanes<16 hold K {0..7,16..23},
// lanes>=16 hold K {8..15,24..31}.  B-frag (16-bit B 32x16): lane%16 = N col,
// 16 elements = K in A's order (stride DOUT in W's row-major storage).
// Q is pre-scaled by 1/sqrt(64); K is stored transposed [B][64][S].
// ---------------------------------------------------------------------------
__global__ __launch_bounds__(128)
void qkv_proj_kernel(const float* __restrict__ Xk, const float* __restrict__ Xv,
                     const float* __restrict__ Xq, const float* __restrict__ Wq,
                     const float* __restrict__ Wk, const float* __restrict__ Wv,
                     __bf16* __restrict__ Qb, __bf16* __restrict__ Ktg,
                     __bf16* __restrict__ Vb) {
  const int mode = blockIdx.y;  // 0=Q, 1=K, 2=V
  const float* __restrict__ X = (mode == 0) ? Xq : (mode == 1) ? Xk : Xv;
  const float* __restrict__ W = (mode == 0) ? Wq : (mode == 1) ? Wk : Wv;

  const int lane = threadIdx.x & 31;
  const int wv   = threadIdx.x >> 5;        // 4 waves -> 4 col groups of 16
  const int l15  = lane & 15;
  const int off  = (lane < 16) ? 0 : 8;     // A-frag K half select
  const int hi8  = (lane < 16) ? 0 : 8;     // C-frag row half select
  const int row0 = blockIdx.x * 16;         // global row in [0, B*S)
  const int cb   = wv * 16;

  const float* __restrict__ Xr = X + (size_t)(row0 + l15) * DIN;

  v8f acc = {};
  #pragma unroll 4
  for (int k0 = 0; k0 < DIN; k0 += 32) {
    // keep the HBM pipe ahead of the cvt+wmma stage
    __builtin_prefetch(Xr + k0 + 128, 0, 1);
    // A: 16 bf16 per lane, two contiguous 32B f32 segments, converted
    v8f alo = *(const v8f*)(Xr + k0 + off);
    v8f ahi = *(const v8f*)(Xr + k0 + 16 + off);
    v16bf a = cat16(cvt8(alo), cvt8(ahi));
    // B: lane%16 = N column, elements = K {off..off+7, 16+off..23+off}
    const float* __restrict__ Wr = W + (size_t)(k0 + off) * DOUT + cb + l15;
    v16bf b;
    #pragma unroll
    for (int i = 0; i < 8; ++i) { b[i] = (__bf16)Wr[i * DOUT]; b[8 + i] = (__bf16)Wr[(16 + i) * DOUT]; }
    acc = WMMA_BF16(a, b, acc);
  }

  // C/D layout: VGPR j -> row (j + hi8), col = l15
  #pragma unroll
  for (int j = 0; j < 8; ++j) {
    const int r = row0 + j + hi8;
    const int c = cb + l15;
    const float v = acc[j];
    if (mode == 0) {
      Qb[(size_t)r * DOUT + c] = (__bf16)(v * 0.125f);  // fold 1/sqrt(64)
    } else if (mode == 1) {
      const int bi = r >> 11, s = r & (SEQ - 1);
      Ktg[((size_t)bi * DOUT + c) * SEQ + s] = (__bf16)v;  // K transposed
    } else {
      Vb[(size_t)r * DOUT + c] = (__bf16)v;
    }
  }
}

// ---------------------------------------------------------------------------
// Kernel 2: causal flash attention. Block = 4 waves = 64 q rows; each wave
// owns a 16-row q tile. k-loop in blocks of 32 keys staged to LDS via the
// async global->LDS engine (no VGPR round-trip; tracked with ASYNCcnt).
// ---------------------------------------------------------------------------
__global__ __launch_bounds__(128)
void flash_attn_kernel(const __bf16* __restrict__ Qb, const __bf16* __restrict__ Ktg,
                       const __bf16* __restrict__ Vb, float* __restrict__ Out) {
  __shared__ __bf16 Kt[64 * 40];        // [d=64][k=32 +8 pad]  (K^T block)
  __shared__ __bf16 Vs[32 * 72];        // [k=32][d=64 +8 pad]  (V block)
  __shared__ __bf16 Ps[4 * 16 * 40];    // per-wave P relayout scratch

  const int tid  = threadIdx.x;
  const int lane = tid & 31;
  const int wv   = tid >> 5;
  const int l15  = lane & 15;
  const int off  = (lane < 16) ? 0 : 8;
  const int hi8  = (lane < 16) ? 0 : 8;
  const int b    = blockIdx.y;
  const int q0   = blockIdx.x * 64 + wv * 16;   // wave's q tile base (local)
  const int nkb  = 2 * blockIdx.x + 2;          // causal #32-key blocks

  // Q fragments for both 32-wide K-dim steps over d=64; kept in VGPRs
  const __bf16* __restrict__ Qrow = Qb + ((size_t)b * SEQ + q0 + l15) * DOUT;
  const v16bf qa0 = cat16(*(const v8bf*)(Qrow + off),      *(const v8bf*)(Qrow + 16 + off));
  const v16bf qa1 = cat16(*(const v8bf*)(Qrow + 32 + off), *(const v8bf*)(Qrow + 48 + off));

  float m[8], l[8];
  v8f o[4] = {{}, {}, {}, {}};
  #pragma unroll
  for (int j = 0; j < 8; ++j) { m[j] = -__builtin_inff(); l[j] = 0.0f; }

  __bf16* __restrict__ Pw = Ps + wv * (16 * 40);

  for (int kb = 0; kb < nkb; ++kb) {
    const int k0 = kb * 32;

    // ---- async staging: 256 x 16B chunks across 128 lanes (ASYNCcnt) ----
    #pragma unroll
    for (int t = 0; t < 2; ++t) {
      const int ci = tid + t * 128;
      const int d = ci >> 2, p = ci & 3;   // Kt: 64 rows x 4 chunks
      async_copy_b128(Kt + d * 40 + p * 8,
                      Ktg + ((size_t)b * DOUT + d) * SEQ + k0 + p * 8);
      const int s = ci >> 3, p2 = ci & 7;  // Vs: 32 rows x 8 chunks
      async_copy_b128(Vs + s * 72 + p2 * 8,
                      Vb + ((size_t)b * SEQ + k0 + s) * DOUT + p2 * 8);
    }
    wait_asynccnt0();
    __syncthreads();

    if (k0 <= q0 + 15) {   // wave-uniform causal skip (EXEC stays all-1s)
      // ---- scores: S[16q x 32k] = Q(16x64) @ K^T(64x32), two C tiles ----
      v8f s0 = {}, s1 = {};
      #pragma unroll
      for (int st = 0; st < 2; ++st) {
        const __bf16* kcol = Kt + (st * 32 + off) * 40 + l15;  // lane%16 = key col
        v16bf b0, b1;
        #pragma unroll
        for (int i = 0; i < 8; ++i) {
          b0[i] = kcol[i * 40];      b0[8 + i] = kcol[(16 + i) * 40];
          b1[i] = kcol[i * 40 + 16]; b1[8 + i] = kcol[(16 + i) * 40 + 16];
        }
        const v16bf a = st ? qa1 : qa0;
        s0 = WMMA_BF16(a, b0, s0);
        s1 = WMMA_BF16(a, b1, s1);
      }

      // ---- causal mask + online softmax (rows live across 16-lane halves) --
      #pragma unroll
      for (int j = 0; j < 8; ++j) {
        const int q = q0 + j + hi8;
        if (k0 + l15 > q)      s0[j] = -__builtin_inff();
        if (k0 + 16 + l15 > q) s1[j] = -__builtin_inff();
        float mj = fmaxf(s0[j], s1[j]);
        #pragma unroll
        for (int d = 1; d < 16; d <<= 1) mj = fmaxf(mj, __shfl_xor(mj, d, 32));
        const float mn = fmaxf(m[j], mj);
        const float al = __expf(m[j] - mn);   // 0 on first block (m=-inf)
        const float p0 = __expf(s0[j] - mn);
        const float p1 = __expf(s1[j] - mn);
        float rs = p0 + p1;
        #pragma unroll
        for (int d = 1; d < 16; d <<= 1) rs += __shfl_xor(rs, d, 32);
        l[j] = l[j] * al + rs;
        m[j] = mn;
        o[0][j] *= al; o[1][j] *= al; o[2][j] *= al; o[3][j] *= al;
        // spill P (C layout) to LDS for A-layout reload
        Pw[(j + hi8) * 40 + l15]      = (__bf16)p0;
        Pw[(j + hi8) * 40 + 16 + l15] = (__bf16)p1;
      }

      // ---- O += P(16x32) @ V(32x64) ----
      const __bf16* prow = Pw + l15 * 40;
      const v16bf pa = cat16(*(const v8bf*)(prow + off), *(const v8bf*)(prow + 16 + off));
      #pragma unroll
      for (int g = 0; g < 4; ++g) {
        const __bf16* vcol = Vs + off * 72 + g * 16 + l15;   // lane%16 = d col
        v16bf vb;
        #pragma unroll
        for (int i = 0; i < 8; ++i) { vb[i] = vcol[i * 72]; vb[8 + i] = vcol[(16 + i) * 72]; }
        o[g] = WMMA_BF16(pa, vb, o[g]);
      }
    }
    __syncthreads();
  }

  // ---- epilogue: O /= rowsum, store f32 ----
  #pragma unroll
  for (int j = 0; j < 8; ++j) {
    const float inv = 1.0f / l[j];
    float* __restrict__ orow = Out + ((size_t)b * SEQ + q0 + j + hi8) * DOUT;
    #pragma unroll
    for (int g = 0; g < 4; ++g) orow[g * 16 + l15] = o[g][j] * inv;
  }
}

// ---------------------------------------------------------------------------
extern "C" void kernel_launch(void* const* d_in, const int* in_sizes, int n_in,
                              void* d_out, int out_size, void* d_ws, size_t ws_size,
                              hipStream_t stream) {
  (void)in_sizes; (void)n_in; (void)out_size; (void)ws_size;
  const float* Xk = (const float*)d_in[0];  // inputs_for_keys
  const float* Xv = (const float*)d_in[1];  // inputs_for_values
  const float* Xq = (const float*)d_in[2];  // inputs_for_queries
  const float* Wq = (const float*)d_in[3];
  const float* Wk = (const float*)d_in[4];
  const float* Wv = (const float*)d_in[5];

  __bf16* Qb  = (__bf16*)d_ws;                       // [B*S][64]
  __bf16* Ktg = Qb  + (size_t)NROWS * DOUT;          // [B][64][S] (transposed)
  __bf16* Vb  = Ktg + (size_t)NROWS * DOUT;          // [B*S][64]

  qkv_proj_kernel<<<dim3(NROWS / 16, 3), 128, 0, stream>>>(
      Xk, Xv, Xq, Wq, Wk, Wv, Qb, Ktg, Vb);
  flash_attn_kernel<<<dim3(SEQ / 64, BATCH), 128, 0, stream>>>(
      Qb, Ktg, Vb, (float*)d_out);
}
